// DisKT_4346506903933
// MI455X (gfx1250) — compile-verified, matrix operand order checked by reading, not run on hardware
//
#include <hip/hip_runtime.h>
#include <hip/hip_bf16.h>

typedef __attribute__((ext_vector_type(16))) _Float16 v16h;
typedef __attribute__((ext_vector_type(8)))  _Float16 v8h;
typedef __attribute__((ext_vector_type(8)))  float    v8f;

#define B_   16
#define S_   1024
#define D_   512
#define H_   8
#define DK_  64

// LDS row strides in halves (padded, 16B-aligned rows)
#define KSTR 72   // K tile: 32 keys x 64 dims (+8 pad)
#define VSTR 40   // V^T tile: 64 dims x 32 keys (+8 pad)
#define ESTR 40   // e scratch: 16 rows x 32 keys (+8 pad)

__device__ __forceinline__ v8h cvt8(float4 a, float4 b) {
    union { v8h v; _Float16 h[8]; } u;
    u.h[0] = (_Float16)a.x; u.h[1] = (_Float16)a.y;
    u.h[2] = (_Float16)a.z; u.h[3] = (_Float16)a.w;
    u.h[4] = (_Float16)b.x; u.h[5] = (_Float16)b.y;
    u.h[6] = (_Float16)b.z; u.h[7] = (_Float16)b.w;
    return u.v;
}

// A/B fragment: 8 contiguous halves at p0, 8 more at p0+16 halves.
__device__ __forceinline__ v16h frag16(const _Float16* p0) {
    union { v16h v; v8h h[2]; } u;
    u.h[0] = *(const v8h*)(p0);
    u.h[1] = *(const v8h*)(p0 + 16);
    return u.v;
}

__global__ __launch_bounds__(256, 1)
void dual_attn_dsoftmax(const float* __restrict__ Q,  const float* __restrict__ K,
                        const float* __restrict__ V1, const float* __restrict__ V2,
                        const int*   __restrict__ CM,
                        float* __restrict__ O1, float* __restrict__ O2)
{
    __shared__ __align__(16) _Float16 ldsK [2][32 * KSTR];
    __shared__ __align__(16) _Float16 ldsV1[2][64 * VSTR];
    __shared__ __align__(16) _Float16 ldsV2[2][64 * VSTR];
    __shared__ __align__(16) _Float16 ldsE [8][16 * ESTR];
    __shared__ float ldsCM[S_];

    const int tid  = threadIdx.x;
    const int wave = tid >> 5;
    const int lane = tid & 31;
    const int hs   = lane >> 4;     // half-wave select (K-dim split of A/B frags)
    const int ln   = lane & 15;     // row (A) / col (B,C) index

    const int bh = blockIdx.x;      // 0..127
    const int b  = bh >> 3;
    const int h  = bh & 7;
    const int tq = blockIdx.y * 8 + wave;  // this wave's query tile
    const int q0 = tq * 16;

    const size_t headOff = (size_t)b * S_ * D_ + (size_t)h * DK_;

    // cooperative-loader coordinates: 8 threads per key row, 8 dims each
    const int kr = tid >> 3;
    const int cb = (tid & 7) * 8;

    // ---- stage counter-attention mask row for this batch (1.0 = masked) ----
    {
        const int i0 = tid * 4;
        #pragma unroll
        for (int i = 0; i < 4; ++i)
            ldsCM[i0 + i] = (CM[(size_t)b * S_ + i0 + i] == 1) ? 1.0f : 0.0f;
    }

    // ---- Q fragments: 16x64 f16 = two 16x32 A fragments, in registers ----
    v16h aQ[2];
    #pragma unroll
    for (int f = 0; f < 2; ++f) {
        const float* qr = Q + headOff + (size_t)(q0 + ln) * D_ + f * 32 + hs * 8;
        float4 x0 = *(const float4*)(qr);
        float4 x1 = *(const float4*)(qr + 4);
        float4 y0 = *(const float4*)(qr + 16);
        float4 y1 = *(const float4*)(qr + 20);
        union { v16h v; v8h p[2]; } u;
        u.p[0] = cvt8(x0, x1);
        u.p[1] = cvt8(y0, y1);
        aQ[f] = u.v;
    }

    // =====================================================================
    // Pass A: softmax-1 row statistics (m, Z1) over causal keys
    //   double-buffered LDS K tiles, register-staged next-tile loads
    // =====================================================================
    float mrow[8], zrow[8];
    #pragma unroll
    for (int r = 0; r < 8; ++r) { mrow[r] = -1e30f; zrow[r] = 0.0f; }

    const int npA = 4 * blockIdx.y + 4;     // key-tile pairs covering tiles 0..tq_max

    {   // prologue: pair 0 -> buffer 0
        const size_t ro = headOff + (size_t)kr * D_ + cb;
        float4 x0 = *(const float4*)(K + ro);
        float4 x1 = *(const float4*)(K + ro + 4);
        *(v8h*)&ldsK[0][kr * KSTR + cb] = cvt8(x0, x1);
    }
    __syncthreads();

    for (int jj = 0; jj < npA; ++jj) {
        const int  cur  = jj & 1;
        const bool more = (jj + 1 < npA);

        // stage next pair's K rows in registers (latency hidden by compute)
        float4 nx0, nx1;
        if (more) {
            const size_t ro = headOff + (size_t)((jj + 1) * 32 + kr) * D_ + cb;
            nx0 = *(const float4*)(K + ro);
            nx1 = *(const float4*)(K + ro + 4);
        }
        if (jj + 2 < npA)
            __builtin_prefetch(K + headOff + (size_t)((jj + 2) * 32 + kr) * D_ + cb, 0, 1);

        #pragma unroll
        for (int st = 0; st < 2; ++st) {
            const int j = jj * 2 + st;
            if (j > tq) continue;           // wave-uniform skip
            v16h bK0 = frag16(&ldsK[cur][(st * 16 + ln) * KSTR + 0  + hs * 8]);
            v16h bK1 = frag16(&ldsK[cur][(st * 16 + ln) * KSTR + 32 + hs * 8]);
            v8f c = {};
            c = __builtin_amdgcn_wmma_f32_16x16x32_f16(false, aQ[0], false, bK0,
                                                       (short)0, c, false, false);
            c = __builtin_amdgcn_wmma_f32_16x16x32_f16(false, aQ[1], false, bK1,
                                                       (short)0, c, false, false);
            const int key = j * 16 + ln;
            #pragma unroll
            for (int r = 0; r < 8; ++r) {
                const int  qg      = q0 + r + hs * 8;
                const bool allowed = key < qg;                   // strict causal
                float v  = c[r] * 0.125f;                        // 1/sqrt(64)
                float vm = allowed ? v : -1e30f;                 // cndmask
                float nm = fmaxf(mrow[r], vm);
                float ea = __expf(v - nm);
                zrow[r]  = zrow[r] * __expf(mrow[r] - nm) + (allowed ? ea : 0.0f);
                mrow[r]  = nm;
            }
        }

        if (more)
            *(v8h*)&ldsK[cur ^ 1][kr * KSTR + cb] = cvt8(nx0, nx1);
        __syncthreads();
    }

    // combine per-lane partial (m,z) across the 16-lane half-wave
    #pragma unroll
    for (int r = 0; r < 8; ++r) {
        float m = mrow[r], z = zrow[r];
        #pragma unroll
        for (int off = 1; off < 16; off <<= 1) {
            float mo = __shfl_xor(m, off, 32);
            float zo = __shfl_xor(z, off, 32);
            float nm = fmaxf(m, mo);
            z = z * __expf(m - nm) + zo * __expf(mo - nm);
            m = nm;
        }
        mrow[r] = m;
        zrow[r] = (z > 0.0f) ? (1.0f / z) : 0.0f;   // invZ1 (row 0 -> 0)
    }

    // =====================================================================
    // Pass B: e = exp(p) over ALL keys, Z2, and e @ V1 / e @ V2
    //   double-buffered K/V tiles, register-staged next-tile loads
    // =====================================================================
    v8f acc1[4] = {};
    v8f acc2[4] = {};
    float z2[8] = {0.f, 0.f, 0.f, 0.f, 0.f, 0.f, 0.f, 0.f};

    {   // prologue: pair 0 -> buffer 0 (K row-major f16, V transposed f16)
        const size_t ro = headOff + (size_t)kr * D_ + cb;
        float4 x0 = *(const float4*)(K  + ro);
        float4 x1 = *(const float4*)(K  + ro + 4);
        float4 a0 = *(const float4*)(V1 + ro);
        float4 a1 = *(const float4*)(V1 + ro + 4);
        float4 b0 = *(const float4*)(V2 + ro);
        float4 b1 = *(const float4*)(V2 + ro + 4);
        *(v8h*)&ldsK[0][kr * KSTR + cb] = cvt8(x0, x1);
        float va[8] = {a0.x, a0.y, a0.z, a0.w, a1.x, a1.y, a1.z, a1.w};
        float vb[8] = {b0.x, b0.y, b0.z, b0.w, b1.x, b1.y, b1.z, b1.w};
        #pragma unroll
        for (int i = 0; i < 8; ++i) {
            ldsV1[0][(cb + i) * VSTR + kr] = (_Float16)va[i];
            ldsV2[0][(cb + i) * VSTR + kr] = (_Float16)vb[i];
        }
    }
    __syncthreads();

    for (int jj = 0; jj < 32; ++jj) {
        const int  cur  = jj & 1;
        const bool more = (jj + 1 < 32);

        // stage next pair in registers
        float4 nk0, nk1, na0, na1, nb0, nb1;
        if (more) {
            const size_t ro = headOff + (size_t)((jj + 1) * 32 + kr) * D_ + cb;
            nk0 = *(const float4*)(K  + ro);
            nk1 = *(const float4*)(K  + ro + 4);
            na0 = *(const float4*)(V1 + ro);
            na1 = *(const float4*)(V1 + ro + 4);
            nb0 = *(const float4*)(V2 + ro);
            nb1 = *(const float4*)(V2 + ro + 4);
        }
        if (jj + 2 < 32) {
            const size_t rp = headOff + (size_t)((jj + 2) * 32 + kr) * D_ + cb;
            __builtin_prefetch(K  + rp, 0, 1);
            __builtin_prefetch(V1 + rp, 0, 1);
            __builtin_prefetch(V2 + rp, 0, 1);
        }

        // scores -> p -> e for both 16-key sub-tiles; stage e in A-layout scratch
        #pragma unroll
        for (int st = 0; st < 2; ++st) {
            const int j = jj * 2 + st;
            v16h bK0 = frag16(&ldsK[cur][(st * 16 + ln) * KSTR + 0  + hs * 8]);
            v16h bK1 = frag16(&ldsK[cur][(st * 16 + ln) * KSTR + 32 + hs * 8]);
            v8f c = {};
            c = __builtin_amdgcn_wmma_f32_16x16x32_f16(false, aQ[0], false, bK0,
                                                       (short)0, c, false, false);
            c = __builtin_amdgcn_wmma_f32_16x16x32_f16(false, aQ[1], false, bK1,
                                                       (short)0, c, false, false);
            const int  key     = j * 16 + ln;
            const bool notmask = (ldsCM[key] == 0.0f);
            #pragma unroll
            for (int r = 0; r < 8; ++r) {
                const int  rowM = r + hs * 8;
                const bool sel  = (key < q0 + rowM) && notmask;  // causal & counter
                float pe = __expf(c[r] * 0.125f - mrow[r]) * zrow[r];
                float p  = sel ? pe : 0.0f;                      // cndmask, no branch
                float e  = __expf(p);                            // softmax-2 numerator
                z2[r] += e;
                ldsE[wave][rowM * ESTR + st * 16 + ln] = (_Float16)e;
            }
        }
        // wave-internal cross-lane LDS dependency (C-layout -> A-layout)
        asm volatile("s_wait_dscnt 0" ::: "memory");

        v16h eA = frag16(&ldsE[wave][ln * ESTR + hs * 8]);   // 16x32 A fragment
        #pragma unroll
        for (int dc = 0; dc < 4; ++dc) {
            v16h bv1 = frag16(&ldsV1[cur][(dc * 16 + ln) * VSTR + hs * 8]);
            v16h bv2 = frag16(&ldsV2[cur][(dc * 16 + ln) * VSTR + hs * 8]);
            acc1[dc] = __builtin_amdgcn_wmma_f32_16x16x32_f16(false, eA, false, bv1,
                                                              (short)0, acc1[dc], false, false);
            acc2[dc] = __builtin_amdgcn_wmma_f32_16x16x32_f16(false, eA, false, bv2,
                                                              (short)0, acc2[dc], false, false);
        }

        // commit staged next pair to the other buffer
        if (more) {
            *(v8h*)&ldsK[cur ^ 1][kr * KSTR + cb] = cvt8(nk0, nk1);
            float va[8] = {na0.x, na0.y, na0.z, na0.w, na1.x, na1.y, na1.z, na1.w};
            float vb[8] = {nb0.x, nb0.y, nb0.z, nb0.w, nb1.x, nb1.y, nb1.z, nb1.w};
            #pragma unroll
            for (int i = 0; i < 8; ++i) {
                ldsV1[cur ^ 1][(cb + i) * VSTR + kr] = (_Float16)va[i];
                ldsV2[cur ^ 1][(cb + i) * VSTR + kr] = (_Float16)vb[i];
            }
        }
        __syncthreads();
    }

    // ---- softmax-2 denominator + epilogue ----
    #pragma unroll
    for (int r = 0; r < 8; ++r) {
        float z = z2[r];
        #pragma unroll
        for (int off = 1; off < 16; off <<= 1)
            z += __shfl_xor(z, off, 32);
        z2[r] = 1.0f / z;
    }

    #pragma unroll
    for (int dc = 0; dc < 4; ++dc) {
        #pragma unroll
        for (int r = 0; r < 8; ++r) {
            const int rowM = r + hs * 8;
            const int qg   = q0 + rowM;
            float o1 = acc1[dc][r] * z2[r];
            float o2 = acc2[dc][r] * z2[r];
            if (qg == 0) { o1 = 0.0f; o2 = 0.0f; }   // reference zeroes q-row 0
            const size_t off = headOff + (size_t)qg * D_ + dc * 16 + ln;
            O1[off] = o1;
            O2[off] = o2;
        }
    }
}

extern "C" void kernel_launch(void* const* d_in, const int* in_sizes, int n_in,
                              void* d_out, int out_size, void* d_ws, size_t ws_size,
                              hipStream_t stream) {
    (void)in_sizes; (void)n_in; (void)d_ws; (void)ws_size; (void)out_size;
    const float* q  = (const float*)d_in[0];
    const float* k  = (const float*)d_in[1];
    const float* v1 = (const float*)d_in[2];
    const float* v2 = (const float*)d_in[3];
    const int*   cm = (const int*)d_in[4];
    float* o1 = (float*)d_out;
    float* o2 = o1 + (size_t)B_ * S_ * D_;

    dim3 grid(B_ * H_, S_ / (16 * 8));   // (128, 8)
    dim3 block(256);
    dual_attn_dsoftmax<<<grid, block, 0, stream>>>(q, k, v1, v2, cm, o1, o2);
}
